// R_Net_48498770706909
// MI455X (gfx1250) — matrix operand order repaired and checked by reading, compiled, not run on hardware
//
#include <hip/hip_runtime.h>
#include <hip/hip_bf16.h>
#include <cstddef>

// ---------------------------------------------------------------------------
// Types / WMMA helpers (gfx1250 wave32, V_WMMA_F32_16X16X32_F16)
// ---------------------------------------------------------------------------
typedef __attribute__((ext_vector_type(16))) _Float16 v16h;
typedef __attribute__((ext_vector_type(8)))  _Float16 h8v;
typedef __attribute__((ext_vector_type(8)))  float    v8f;

__device__ inline v8f wmma16(v16h a, v16h b, v8f c) {
  // (neg_a, A, neg_b, B, c_mod, C, reuse_a, reuse_b)
  return __builtin_amdgcn_wmma_f32_16x16x32_f16(false, a, false, b, (short)0, c,
                                                false, false);
}

__device__ inline float sigf(float x) { return 1.f / (1.f + __expf(-x)); }

// A fragment: 16x32 f16, row-major source (lda = padded K, mult of 32).
// Lanes 0-15 hold row M=lane, K = kb+{0..7,16..23}; lanes 16-31: K = kb+{8..15,24..31}.
__device__ inline v16h load_a_frag(const _Float16* A, int lda, int m0, int kb, int lane) {
  int r = lane & 15, hi = (lane >> 4) & 1;
  const _Float16* p = A + (size_t)(m0 + r) * lda + kb + hi * 8;
  h8v lo = *(const h8v*)p;
  h8v hh = *(const h8v*)(p + 16);
  v16h a;
#pragma unroll
  for (int i = 0; i < 8; ++i) { a[i] = lo[i]; a[8 + i] = hh[i]; }
  return a;
}

// B fragment: 32x16 f16 from W row-major [Npad, Kpad] (so B = W^T col tile).
// Lane n=lane&15; low lanes K=kb+{0..7,16..23}, high lanes K=kb+{8..15,24..31}.
__device__ inline v16h load_b_frag(const _Float16* W, int ldw, int n0, int kb, int lane) {
  int n = n0 + (lane & 15), hi = (lane >> 4) & 1;
  const _Float16* p = W + (size_t)n * ldw + kb + hi * 8;
  h8v lo = *(const h8v*)p;
  h8v hh = *(const h8v*)(p + 16);
  v16h b;
#pragma unroll
  for (int i = 0; i < 8; ++i) { b[i] = lo[i]; b[8 + i] = hh[i]; }
  return b;
}

// In-block cooperative GEMM: C[M,N](f32) = A[M,Kp] @ W[N,Kp]^T (+bias[n]).
// One 16x16 tile per wave iteration. No barriers inside (caller syncs).
__device__ inline void block_gemm_f32(const _Float16* A, int lda,
                                      const _Float16* W, int ldw,
                                      float* C, int ldc, const float* bias,
                                      int M, int N, int Kp,
                                      int lane, int wave, int nwaves) {
  int mt = (M + 15) >> 4, nt = (N + 15) >> 4;
  for (int tile = wave; tile < mt * nt; tile += nwaves) {
    int m0 = (tile / nt) << 4, n0 = (tile % nt) << 4;
    v8f acc = {};
    for (int kb = 0; kb < Kp; kb += 32)
      acc = wmma16(load_a_frag(A, lda, m0, kb, lane),
                   load_b_frag(W, ldw, n0, kb, lane), acc);
    int cn = n0 + (lane & 15);
    int mh = ((lane >> 4) & 1) << 3;
    if (cn < N) {
      float bv = bias ? bias[cn] : 0.f;
#pragma unroll
      for (int r = 0; r < 8; ++r) {
        int mm = m0 + mh + r;
        if (mm < M) C[(size_t)mm * ldc + cn] = acc[r] + bv;
      }
    }
  }
}

// Same, but stores sigmoid(acc) as f16 (used for the attention gates, no bias).
__device__ inline void block_gemm_sig_f16(const _Float16* A, int lda,
                                          const _Float16* W, int ldw,
                                          _Float16* C, int ldc,
                                          int M, int N, int Kp,
                                          int lane, int wave, int nwaves) {
  int mt = (M + 15) >> 4, nt = (N + 15) >> 4;
  for (int tile = wave; tile < mt * nt; tile += nwaves) {
    int m0 = (tile / nt) << 4, n0 = (tile % nt) << 4;
    v8f acc = {};
    for (int kb = 0; kb < Kp; kb += 32)
      acc = wmma16(load_a_frag(A, lda, m0, kb, lane),
                   load_b_frag(W, ldw, n0, kb, lane), acc);
    int cn = n0 + (lane & 15);
    int mh = ((lane >> 4) & 1) << 3;
    if (cn < N) {
#pragma unroll
      for (int r = 0; r < 8; ++r) {
        int mm = m0 + mh + r;
        if (mm < M) C[(size_t)mm * ldc + cn] = (_Float16)sigf(acc[r]);
      }
    }
  }
}

// ---------------------------------------------------------------------------
// Utility kernels
// ---------------------------------------------------------------------------
__global__ void k_pack_f16(const float* __restrict__ src, int srcld, int rows, int cols,
                           _Float16* __restrict__ dst, int dstld, int dstRows) {
  int total = dstRows * dstld;
  for (int i = blockIdx.x * blockDim.x + threadIdx.x; i < total;
       i += gridDim.x * blockDim.x) {
    int r = i / dstld, c = i % dstld;
    float v = (r < rows && c < cols) ? src[(size_t)r * srcld + c] : 0.f;
    dst[i] = (_Float16)v;
  }
}

__global__ void k_zero_f16(_Float16* p, int n) {
  for (int i = blockIdx.x * blockDim.x + threadIdx.x; i < n;
       i += gridDim.x * blockDim.x)
    p[i] = (_Float16)0.f;
}

// Grid-wide batched GEMM: C = A @ W^T (+bias). One 16x16 tile per wave.
__global__ void k_gemm_wmma(const _Float16* __restrict__ A, int lda,
                            const _Float16* __restrict__ W, int ldw,
                            float* __restrict__ C, int ldc,
                            const float* __restrict__ bias,
                            int M, int N, int Kp) {
  int lane = threadIdx.x & 31;
  int gw = blockIdx.x * (blockDim.x >> 5) + (threadIdx.x >> 5);
  int nw = gridDim.x * (blockDim.x >> 5);
  int mt = (M + 15) >> 4, nt = (N + 15) >> 4;
  for (int tile = gw; tile < mt * nt; tile += nw) {
    int m0 = (tile / nt) << 4, n0 = (tile % nt) << 4;
    v8f acc = {};
    for (int kb = 0; kb < Kp; kb += 32) {
      acc = wmma16(load_a_frag(A, lda, m0, kb, lane),
                   load_b_frag(W, ldw, n0, kb, lane), acc);
    }
    int cn = n0 + (lane & 15);
    int mh = ((lane >> 4) & 1) << 3;
    if (cn < N) {
      float bv = bias ? bias[cn] : 0.f;
#pragma unroll
      for (int r = 0; r < 8; ++r) {
        int mm = m0 + mh + r;
        if (mm < M) C[(size_t)mm * ldc + cn] = acc[r] + bv;
      }
    }
  }
}

// ---------------------------------------------------------------------------
// Char-level GRU: each block owns 16 sequences, loops over CL=16 steps.
// Both gi and gh GEMMs done in-block with WMMA; writes final hidden (f16)
// directly into the reader's layer-0 input at column `outoff`.
// ---------------------------------------------------------------------------
__global__ void k_char_gru(const float* __restrict__ X,          // [Nseq,16,300]
                           const _Float16* __restrict__ Wih,     // [240,320]
                           const _Float16* __restrict__ Whh,     // [240,96]
                           const float* __restrict__ bih,
                           const float* __restrict__ bhh,
                           _Float16* __restrict__ Hout, int outld, int outoff,
                           int reverse) {
  __shared__ _Float16 sX[16 * 320];
  __shared__ _Float16 sH[16 * 96];
  __shared__ float sGi[16 * 240];
  __shared__ float sGh[16 * 240];
  const int tid = threadIdx.x, lane = tid & 31, wave = tid >> 5, nw = blockDim.x >> 5;
  const int seq0 = blockIdx.x * 16;
  for (int i = tid; i < 16 * 320; i += blockDim.x) sX[i] = (_Float16)0.f;
  for (int i = tid; i < 16 * 96; i += blockDim.x) sH[i] = (_Float16)0.f;
  __syncthreads();
  for (int s = 0; s < 16; ++s) {
    int t = reverse ? (15 - s) : s;
    for (int i = tid; i < 16 * 300; i += blockDim.x) {
      int m = i / 300, c = i % 300;
      sX[m * 320 + c] = (_Float16)X[((size_t)(seq0 + m) * 16 + t) * 300 + c];
    }
    __syncthreads();
    block_gemm_f32(sX, 320, Wih, 320, sGi, 240, bih, 16, 225, 320, lane, wave, nw);
    block_gemm_f32(sH, 96, Whh, 96, sGh, 240, bhh, 16, 225, 96, lane, wave, nw);
    __syncthreads();
    for (int i = tid; i < 16 * 75; i += blockDim.x) {
      int m = i / 75, j = i % 75;
      float hp = (float)sH[m * 96 + j];
      float r = sigf(sGi[m * 240 + j] + sGh[m * 240 + j]);
      float z = sigf(sGi[m * 240 + 75 + j] + sGh[m * 240 + 75 + j]);
      float n = tanhf(sGi[m * 240 + 150 + j] + r * sGh[m * 240 + 150 + j]);
      sH[m * 96 + j] = (_Float16)((1.f - z) * n + z * hp);
    }
    __syncthreads();
  }
  for (int i = tid; i < 16 * 75; i += blockDim.x) {
    int m = i / 75, j = i % 75;
    Hout[(size_t)(seq0 + m) * outld + outoff + j] = sH[m * 96 + j];
  }
}

// ---------------------------------------------------------------------------
// Reader recurrence: gi precomputed (incl. bih) by k_gemm_wmma for the whole
// sequence; this kernel runs the serial h-chain (gh via WMMA). Each block
// owns 16 batch rows. Writes f32 into the concat output and f16 into the
// next layer's padded input.
// ---------------------------------------------------------------------------
__global__ void k_gru_time(const float* __restrict__ Gi,       // [(32*T),240]
                           const _Float16* __restrict__ Whh,   // [240,96]
                           const float* __restrict__ bhh,
                           float* __restrict__ Y, int ldy, int yoff,
                           _Float16* __restrict__ Xn, int ldxn, int xoff,
                           int T, int reverse) {
  __shared__ _Float16 sH[16 * 96];
  __shared__ float sGh[16 * 240];
  const int tid = threadIdx.x, lane = tid & 31, wave = tid >> 5, nw = blockDim.x >> 5;
  const int b0 = blockIdx.x * 16;
  for (int i = tid; i < 16 * 96; i += blockDim.x) sH[i] = (_Float16)0.f;
  __syncthreads();
  for (int s = 0; s < T; ++s) {
    int t = reverse ? (T - 1 - s) : s;
    block_gemm_f32(sH, 96, Whh, 96, sGh, 240, bhh, 16, 225, 96, lane, wave, nw);
    __syncthreads();
    for (int i = tid; i < 16 * 75; i += blockDim.x) {
      int m = i / 75, j = i % 75;
      size_t row = (size_t)(b0 + m) * T + t;
      float hp = (float)sH[m * 96 + j];
      float r = sigf(Gi[row * 240 + j] + sGh[m * 240 + j]);
      float z = sigf(Gi[row * 240 + 75 + j] + sGh[m * 240 + 75 + j]);
      float n = tanhf(Gi[row * 240 + 150 + j] + r * sGh[m * 240 + 150 + j]);
      float hn = (1.f - z) * n + z * hp;
      sH[m * 96 + j] = (_Float16)hn;
      Y[row * ldy + yoff + j] = hn;
      Xn[row * ldxn + xoff + j] = (_Float16)hn;
    }
    __syncthreads();
  }
}

// ---------------------------------------------------------------------------
// QPP: persistent workgroup, one per direction. State v[32,75] in LDS.
// Per step: attention over QL=50, gated 900-wide input, GRU cell.
// Gate / gi / gh / WPv-proj GEMMs all in-block WMMA.
// ---------------------------------------------------------------------------
__global__ void k_qpp(const float* __restrict__ uP,      // [32*400,450]
                      const float* __restrict__ uPWPu,   // [32*400,80]
                      const float* __restrict__ WQuQ,    // [32*50,80]
                      const float* __restrict__ uQ,      // [32*50,450]
                      const float* __restrict__ vvec,    // [75]
                      const _Float16* __restrict__ Wg1p, // [912,928]
                      const _Float16* __restrict__ WPvp, // [80,96]
                      const _Float16* __restrict__ Wihp, // [240,928]
                      const _Float16* __restrict__ Whhp, // [240,96]
                      const float* __restrict__ bih, const float* __restrict__ bhh,
                      float* __restrict__ vP, int voff, int reverse) {
  extern __shared__ char smem[];
  _Float16* sInp = (_Float16*)smem;            // 32*928
  _Float16* sG = sInp + 32 * 928;              // 32*928
  _Float16* sV = sG + 32 * 928;                // 32*96
  float* sProj = (float*)(sV + 32 * 96);       // 32*80
  float* sS = sProj + 32 * 80;                 // 32*64
  float* sGi = sS + 32 * 64;                   // 32*240
  float* sGh = sGi + 32 * 240;                 // 32*240
  const int tid = threadIdx.x, nth = blockDim.x;
  const int lane = tid & 31, wave = tid >> 5, nw = blockDim.x >> 5;
  for (int i = tid; i < 32 * 928; i += nth) { sInp[i] = (_Float16)0.f; sG[i] = (_Float16)0.f; }
  for (int i = tid; i < 32 * 96; i += nth) sV[i] = (_Float16)0.f;
  __syncthreads();
  for (int s = 0; s < 400; ++s) {
    int t = reverse ? (399 - s) : s;
    for (int i = tid; i < 32 * 450; i += nth) {
      int m = i / 450, c = i % 450;
      sInp[m * 928 + c] = (_Float16)uP[((size_t)m * 400 + t) * 450 + c];
    }
    __syncthreads();
    block_gemm_f32(sV, 96, WPvp, 96, sProj, 80, nullptr, 32, 75, 96, lane, wave, nw);
    __syncthreads();
    for (int i = tid; i < 32 * 50; i += nth) {
      int m = i / 50, q = i % 50;
      const float* wq = WQuQ + ((size_t)m * 50 + q) * 80;
      const float* wp = uPWPu + ((size_t)m * 400 + t) * 80;
      float acc = 0.f;
      for (int j = 0; j < 75; ++j)
        acc += vvec[j] * tanhf(wq[j] + wp[j] + sProj[m * 80 + j]);
      sS[m * 64 + q] = acc;
    }
    __syncthreads();
    for (int m = tid; m < 32; m += nth) {
      float mx = -1e30f;
      for (int q = 0; q < 50; ++q) mx = fmaxf(mx, sS[m * 64 + q]);
      float sum = 0.f;
      for (int q = 0; q < 50; ++q) { float e = __expf(sS[m * 64 + q] - mx); sS[m * 64 + q] = e; sum += e; }
      float inv = 1.f / sum;
      for (int q = 0; q < 50; ++q) sS[m * 64 + q] *= inv;
    }
    __syncthreads();
    for (int i = tid; i < 32 * 450; i += nth) {
      int m = i / 450, d = i % 450;
      float acc = 0.f;
      for (int q = 0; q < 50; ++q)
        acc += sS[m * 64 + q] * uQ[((size_t)m * 50 + q) * 450 + d];
      sInp[m * 928 + 450 + d] = (_Float16)acc;
    }
    __syncthreads();
    block_gemm_sig_f16(sInp, 928, Wg1p, 928, sG, 928, 32, 900, 928, lane, wave, nw);
    __syncthreads();
    for (int i = tid; i < 32 * 900; i += nth) {
      int m = i / 900, c = i % 900;
      sInp[m * 928 + c] = (_Float16)((float)sInp[m * 928 + c] * (float)sG[m * 928 + c]);
    }
    __syncthreads();
    block_gemm_f32(sInp, 928, Wihp, 928, sGi, 240, bih, 32, 225, 928, lane, wave, nw);
    block_gemm_f32(sV, 96, Whhp, 96, sGh, 240, bhh, 32, 225, 96, lane, wave, nw);
    __syncthreads();
    for (int i = tid; i < 32 * 75; i += nth) {
      int m = i / 75, j = i % 75;
      float hp = (float)sV[m * 96 + j];
      float r = sigf(sGi[m * 240 + j] + sGh[m * 240 + j]);
      float z = sigf(sGi[m * 240 + 75 + j] + sGh[m * 240 + 75 + j]);
      float n = tanhf(sGi[m * 240 + 150 + j] + r * sGh[m * 240 + 150 + j]);
      float hn = (1.f - z) * n + z * hp;
      sV[m * 96 + j] = (_Float16)hn;
      vP[((size_t)m * 400 + t) * 150 + voff + j] = hn;
    }
    __syncthreads();
  }
}

// ---------------------------------------------------------------------------
// Self-matching: same structure, attention over PL=400, 300-wide gated input.
// ---------------------------------------------------------------------------
__global__ void k_sm(const float* __restrict__ vPf,      // [32*400,150]
                     const float* __restrict__ WpvP,     // [32*400,80]
                     const float* __restrict__ vvec,
                     const _Float16* __restrict__ Wg2p,  // [304,320]
                     const _Float16* __restrict__ Wp2p,  // [80,160]
                     const _Float16* __restrict__ Wihp,  // [240,320]
                     const _Float16* __restrict__ Whhp,  // [240,96]
                     const float* __restrict__ bih, const float* __restrict__ bhh,
                     float* __restrict__ hP, int hoff, int reverse) {
  extern __shared__ char smem[];
  _Float16* sInp = (_Float16*)smem;        // 32*320
  _Float16* sG = sInp + 32 * 320;          // 32*320
  _Float16* sH = sG + 32 * 320;            // 32*96
  float* sProj = (float*)(sH + 32 * 96);   // 32*80
  float* sGi = sProj + 32 * 80;            // 32*240
  float* sGh = sGi + 32 * 240;             // 32*240
  float* sS = sGh + 32 * 240;              // 32*400
  const int tid = threadIdx.x, nth = blockDim.x;
  const int lane = tid & 31, wave = tid >> 5, nw = blockDim.x >> 5;
  for (int i = tid; i < 32 * 320; i += nth) { sInp[i] = (_Float16)0.f; sG[i] = (_Float16)0.f; }
  for (int i = tid; i < 32 * 96; i += nth) sH[i] = (_Float16)0.f;
  __syncthreads();
  for (int s = 0; s < 400; ++s) {
    int t = reverse ? (399 - s) : s;
    for (int i = tid; i < 32 * 160; i += nth) {
      int m = i / 160, c = i % 160;
      sInp[m * 320 + c] = (c < 150)
          ? (_Float16)vPf[((size_t)m * 400 + t) * 150 + c] : (_Float16)0.f;
    }
    __syncthreads();
    block_gemm_f32(sInp, 320, Wp2p, 160, sProj, 80, nullptr, 32, 75, 160, lane, wave, nw);
    __syncthreads();
    for (int i = tid; i < 32 * 400; i += nth) {
      int m = i / 400, p = i % 400;
      const float* wp = WpvP + ((size_t)m * 400 + p) * 80;
      float acc = 0.f;
      for (int j = 0; j < 75; ++j) acc += vvec[j] * tanhf(wp[j] + sProj[m * 80 + j]);
      sS[m * 400 + p] = acc;
    }
    __syncthreads();
    for (int m = tid; m < 32; m += nth) {
      float mx = -1e30f;
      for (int p = 0; p < 400; ++p) mx = fmaxf(mx, sS[m * 400 + p]);
      float sum = 0.f;
      for (int p = 0; p < 400; ++p) { float e = __expf(sS[m * 400 + p] - mx); sS[m * 400 + p] = e; sum += e; }
      float inv = 1.f / sum;
      for (int p = 0; p < 400; ++p) sS[m * 400 + p] *= inv;
    }
    __syncthreads();
    for (int i = tid; i < 32 * 300; i += nth) {
      int m = i / 300, d = i % 300;
      float acc = 0.f;
      for (int p = 0; p < 400; ++p)
        acc += sS[m * 400 + p] * vPf[((size_t)m * 400 + p) * 150 + d];
      sInp[m * 320 + 150 + d] = (_Float16)acc;
    }
    __syncthreads();
    block_gemm_sig_f16(sInp, 320, Wg2p, 320, sG, 320, 32, 300, 320, lane, wave, nw);
    __syncthreads();
    for (int i = tid; i < 32 * 300; i += nth) {
      int m = i / 300, c = i % 300;
      sInp[m * 320 + c] = (_Float16)((float)sInp[m * 320 + c] * (float)sG[m * 320 + c]);
    }
    __syncthreads();
    block_gemm_f32(sInp, 320, Wihp, 320, sGi, 240, bih, 32, 225, 320, lane, wave, nw);
    block_gemm_f32(sH, 96, Whhp, 96, sGh, 240, bhh, 32, 225, 96, lane, wave, nw);
    __syncthreads();
    for (int i = tid; i < 32 * 75; i += nth) {
      int m = i / 75, j = i % 75;
      float hp = (float)sH[m * 96 + j];
      float r = sigf(sGi[m * 240 + j] + sGh[m * 240 + j]);
      float z = sigf(sGi[m * 240 + 75 + j] + sGh[m * 240 + 75 + j]);
      float n = tanhf(sGi[m * 240 + 150 + j] + r * sGh[m * 240 + 150 + j]);
      float hn = (1.f - z) * n + z * hp;
      sH[m * 96 + j] = (_Float16)hn;
      hP[((size_t)m * 400 + t) * 150 + hoff + j] = hn;
    }
    __syncthreads();
  }
}

// ---------------------------------------------------------------------------
// Pointer network: single workgroup, VALU (tiny stage).
// ---------------------------------------------------------------------------
__global__ void k_pointer(const float* __restrict__ WQuQ,   // [1600,80]
                          const float* __restrict__ uQ,     // [1600,450]
                          const float* __restrict__ hP,     // [12800,150]
                          const float* __restrict__ WPhhP,  // [12800,80]
                          const float* __restrict__ vvec,   // [75]
                          const float* __restrict__ WQv,    // [75,150]
                          const float* __restrict__ VQr,    // [150]
                          const float* __restrict__ Wah,    // [75,450]
                          const float* __restrict__ PNWih,  // [1350,150]
                          const float* __restrict__ PNWhh,  // [1350,450]
                          const float* __restrict__ PNbih,  // [1350]
                          const float* __restrict__ PNbhh,  // [1350]
                          float* __restrict__ out) {        // [2,32,400]
  extern __shared__ char smem[];
  float* sWqv = (float*)smem;          // 80
  float* sHW = sWqv + 80;              // 32*80
  float* sHa = sHW + 32 * 80;          // 32*480
  float* sHa2 = sHa + 32 * 480;        // 32*480
  float* sC1 = sHa2 + 32 * 480;        // 32*304
  float* sS = sC1 + 32 * 304;          // 32*400
  const int tid = threadIdx.x, nth = blockDim.x;
  for (int j = tid; j < 80; j += nth) {
    float acc = 0.f;
    if (j < 75) for (int d = 0; d < 150; ++d) acc += VQr[d] * WQv[j * 150 + d];
    sWqv[j] = acc;
  }
  __syncthreads();
  for (int i = tid; i < 32 * 50; i += nth) {
    int m = i / 50, q = i % 50;
    float acc = 0.f;
    for (int j = 0; j < 75; ++j)
      acc += vvec[j] * tanhf(WQuQ[((size_t)m * 50 + q) * 80 + j] + sWqv[j]);
    sS[m * 400 + q] = acc;
  }
  __syncthreads();
  for (int m = tid; m < 32; m += nth) {
    float mx = -1e30f;
    for (int q = 0; q < 50; ++q) mx = fmaxf(mx, sS[m * 400 + q]);
    float sum = 0.f;
    for (int q = 0; q < 50; ++q) { float e = __expf(sS[m * 400 + q] - mx); sS[m * 400 + q] = e; sum += e; }
    float inv = 1.f / sum;
    for (int q = 0; q < 50; ++q) sS[m * 400 + q] *= inv;
  }
  __syncthreads();
  for (int i = tid; i < 32 * 450; i += nth) {
    int m = i / 450, d = i % 450;
    float a = 0.f;
    for (int q = 0; q < 50; ++q) a += sS[m * 400 + q] * uQ[((size_t)m * 50 + q) * 450 + d];
    sHa[m * 480 + d] = a;
  }
  __syncthreads();
  for (int i = tid; i < 32 * 75; i += nth) {
    int m = i / 75, j = i % 75;
    float a = 0.f;
    for (int d = 0; d < 450; ++d) a += sHa[m * 480 + d] * Wah[j * 450 + d];
    sHW[m * 80 + j] = a;
  }
  __syncthreads();
  for (int i = tid; i < 32 * 400; i += nth) {
    int m = i / 400, p = i % 400;
    float a = 0.f;
    for (int j = 0; j < 75; ++j)
      a += vvec[j] * tanhf(WPhhP[((size_t)m * 400 + p) * 80 + j] + sHW[m * 80 + j]);
    out[m * 400 + p] = a;
    sS[m * 400 + p] = a;
  }
  __syncthreads();
  for (int m = tid; m < 32; m += nth) {
    float mx = -1e30f;
    for (int p = 0; p < 400; ++p) mx = fmaxf(mx, sS[m * 400 + p]);
    float sum = 0.f;
    for (int p = 0; p < 400; ++p) { float e = __expf(sS[m * 400 + p] - mx); sS[m * 400 + p] = e; sum += e; }
    float inv = 1.f / sum;
    for (int p = 0; p < 400; ++p) sS[m * 400 + p] *= inv;
  }
  __syncthreads();
  for (int i = tid; i < 32 * 300; i += nth) {
    int m = i / 300, d = i % 300;
    float a = 0.f;
    for (int p = 0; p < 400; ++p) a += sS[m * 400 + p] * hP[((size_t)m * 400 + p) * 150 + d];
    sC1[m * 304 + d] = a;
  }
  __syncthreads();
  for (int i = tid; i < 32 * 450; i += nth) {
    int m = i / 450, j = i % 450;
    float gir = PNbih[j], giz = PNbih[450 + j], gin = PNbih[900 + j];
    for (int d = 0; d < 150; ++d) {
      float c = sC1[m * 304 + d];
      gir += c * PNWih[(size_t)j * 150 + d];
      giz += c * PNWih[(size_t)(450 + j) * 150 + d];
      gin += c * PNWih[(size_t)(900 + j) * 150 + d];
    }
    float ghr = PNbhh[j], ghz = PNbhh[450 + j], ghn = PNbhh[900 + j];
    for (int d = 0; d < 450; ++d) {
      float h = sHa[m * 480 + d];
      ghr += h * PNWhh[(size_t)j * 450 + d];
      ghz += h * PNWhh[(size_t)(450 + j) * 450 + d];
      ghn += h * PNWhh[(size_t)(900 + j) * 450 + d];
    }
    float r = sigf(gir + ghr), z = sigf(giz + ghz), n = tanhf(gin + r * ghn);
    sHa2[m * 480 + j] = (1.f - z) * n + z * sHa[m * 480 + j];
  }
  __syncthreads();
  for (int i = tid; i < 32 * 75; i += nth) {
    int m = i / 75, j = i % 75;
    float a = 0.f;
    for (int d = 0; d < 450; ++d) a += sHa2[m * 480 + d] * Wah[j * 450 + d];
    sHW[m * 80 + j] = a;
  }
  __syncthreads();
  for (int i = tid; i < 32 * 400; i += nth) {
    int m = i / 400, p = i % 400;
    float a = 0.f;
    for (int j = 0; j < 75; ++j)
      a += vvec[j] * tanhf(WPhhP[((size_t)m * 400 + p) * 80 + j] + sHW[m * 80 + j]);
    out[12800 + m * 400 + p] = a;
  }
}

// ---------------------------------------------------------------------------
// Host orchestration
// ---------------------------------------------------------------------------
extern "C" void kernel_launch(void* const* d_in, const int* in_sizes, int n_in,
                              void* d_out, int out_size, void* d_ws, size_t ws_size,
                              hipStream_t stream) {
  (void)in_sizes; (void)n_in; (void)out_size; (void)ws_size;
  const float* P_emb = (const float*)d_in[0];
  const float* Q_emb = (const float*)d_in[1];
  const float* P_char = (const float*)d_in[2];
  const float* Q_char = (const float*)d_in[3];
  // Flattened param indices (dict order): GRU sets are {Wih,Whh,bih,bhh}.
  const int Pc_f = 4, Pc_b = 8, Qc_f = 12, Qc_b = 16, PrB = 20, QrB = 44;
  const int QPPf = 68, QPPb = 72, SMf = 76, SMb = 80, PNI = 84;
  const int Iv = 88, IWQu = 89, IWPu = 90, IWPv = 91, IWg1 = 92, ISMWp = 93,
            ISMWp2 = 94, IWg2 = 95, IWPh = 96, IWah = 97, IWQv = 98, IVQr = 99;
  auto F = [&](int i) { return (const float*)d_in[i]; };

  char* wsb = (char*)d_ws;
  size_t off = 0;
  auto alloc = [&](size_t bytes) -> void* {
    off = (off + 255) & ~(size_t)255;
    void* p = wsb + off;
    off += bytes;
    return p;
  };
  auto packW = [&](int idx, int N, int K, int Np, int Kp) -> const _Float16* {
    _Float16* dst = (_Float16*)alloc((size_t)Np * Kp * sizeof(_Float16));
    int total = Np * Kp;
    k_pack_f16<<<(total + 255) / 256, 256, 0, stream>>>(F(idx), K, N, K, dst, Kp, Np);
    return dst;
  };

  // ---- pack all weights to padded f16 (resident in L2 thereafter) ----
  const _Float16 *cWih[4], *cWhh[4];
  int cset[4] = {Pc_f, Pc_b, Qc_f, Qc_b};
  for (int i = 0; i < 4; ++i) {
    cWih[i] = packW(cset[i] + 0, 225, 300, 240, 320);
    cWhh[i] = packW(cset[i] + 1, 225, 75, 240, 96);
  }
  const _Float16 *rWih[12], *rWhh[12];
  for (int i = 0; i < 12; ++i) {
    int base = (i < 6) ? (PrB + i * 4) : (QrB + (i - 6) * 4);
    int K = ((i % 6) < 2) ? 450 : 150;
    int Kp = ((i % 6) < 2) ? 480 : 160;
    rWih[i] = packW(base + 0, 225, K, 240, Kp);
    rWhh[i] = packW(base + 1, 225, 75, 240, 96);
  }
  const _Float16* qWih[2] = {packW(QPPf, 225, 900, 240, 928), packW(QPPb, 225, 900, 240, 928)};
  const _Float16* qWhh[2] = {packW(QPPf + 1, 225, 75, 240, 96), packW(QPPb + 1, 225, 75, 240, 96)};
  const _Float16* smWih[2] = {packW(SMf, 225, 300, 240, 320), packW(SMb, 225, 300, 240, 320)};
  const _Float16* smWhh[2] = {packW(SMf + 1, 225, 75, 240, 96), packW(SMb + 1, 225, 75, 240, 96)};
  const _Float16* pWQu = packW(IWQu, 75, 450, 80, 480);
  const _Float16* pWPu = packW(IWPu, 75, 450, 80, 480);
  const _Float16* pWPv = packW(IWPv, 75, 75, 80, 96);
  const _Float16* pWg1 = packW(IWg1, 900, 900, 912, 928);
  const _Float16* pSMWp = packW(ISMWp, 75, 150, 80, 160);
  const _Float16* pSMWp2 = packW(ISMWp2, 75, 150, 80, 160);
  const _Float16* pWg2 = packW(IWg2, 300, 300, 304, 320);
  const _Float16* pWPh = packW(IWPh, 75, 150, 80, 160);

  // ---- activation workspace ----
  _Float16* X0P = (_Float16*)alloc(12800ull * 480 * 2);
  _Float16* X0Q = (_Float16*)alloc(1600ull * 480 * 2);
  _Float16* X1P = (_Float16*)alloc(12800ull * 160 * 2);
  _Float16* X1Q = (_Float16*)alloc(1600ull * 160 * 2);
  float* GiA = (float*)alloc(12800ull * 240 * 4);
  float* GiB = (float*)alloc(12800ull * 240 * 4);
  float* uP = (float*)alloc(12800ull * 450 * 4);
  float* uQ = (float*)alloc(1600ull * 450 * 4);
  _Float16* uPf16 = (_Float16*)alloc(12800ull * 480 * 2);
  _Float16* uQf16 = (_Float16*)alloc(1600ull * 480 * 2);
  float* uPWPu = (float*)alloc(12800ull * 80 * 4);
  float* WQuQ = (float*)alloc(1600ull * 80 * 4);
  float* vPbuf = (float*)alloc(12800ull * 150 * 4);
  _Float16* vPf16 = (_Float16*)alloc(12800ull * 160 * 2);
  float* WpvP = (float*)alloc(12800ull * 80 * 4);
  float* hPbuf = (float*)alloc(12800ull * 150 * 4);
  _Float16* hPf16 = (_Float16*)alloc(12800ull * 160 * 2);
  float* WPhhP = (float*)alloc(12800ull * 80 * 4);

  // ---- embeddings -> layer-0 inputs (zero-pads cols >= 300) ----
  k_pack_f16<<<(12800 * 480 + 255) / 256, 256, 0, stream>>>(P_emb, 300, 12800, 300, X0P, 480, 12800);
  k_pack_f16<<<(1600 * 480 + 255) / 256, 256, 0, stream>>>(Q_emb, 300, 1600, 300, X0Q, 480, 1600);

  // ---- char encoders (final hiddens written at cols 300/375) ----
  k_char_gru<<<800, 512, 0, stream>>>(P_char, cWih[0], cWhh[0], F(Pc_f + 2), F(Pc_f + 3), X0P, 480, 300, 0);
  k_char_gru<<<800, 512, 0, stream>>>(P_char, cWih[1], cWhh[1], F(Pc_b + 2), F(Pc_b + 3), X0P, 480, 375, 1);
  k_char_gru<<<100, 512, 0, stream>>>(Q_char, cWih[2], cWhh[2], F(Qc_f + 2), F(Qc_f + 3), X0Q, 480, 300, 0);
  k_char_gru<<<100, 512, 0, stream>>>(Q_char, cWih[3], cWhh[3], F(Qc_b + 2), F(Qc_b + 3), X0Q, 480, 375, 1);

  k_zero_f16<<<(12800 * 160 + 255) / 256, 256, 0, stream>>>(X1P, 12800 * 160);
  k_zero_f16<<<(1600 * 160 + 255) / 256, 256, 0, stream>>>(X1Q, 1600 * 160);

  // ---- 3-layer bi-GRU readers: big gi GEMM + serial recurrence ----
  for (int pass = 0; pass < 2; ++pass) {
    int T = pass ? 50 : 400;
    int M = 32 * T;
    const _Float16* X0 = pass ? X0Q : X0P;
    _Float16* X1 = pass ? X1Q : X1P;
    float* U = pass ? uQ : uP;
    for (int L = 0; L < 3; ++L) {
      const _Float16* Ain = L ? (const _Float16*)X1 : X0;
      int Kp = L ? 160 : 480;
      int w0 = pass * 6 + L * 2;
      int tiles = ((M + 15) / 16) * 15;
      for (int d = 0; d < 2; ++d) {
        int si = (pass ? QrB : PrB) + (L * 2 + d) * 4;
        k_gemm_wmma<<<(tiles + 7) / 8, 256, 0, stream>>>(
            Ain, Kp, rWih[w0 + d], Kp, d ? GiB : GiA, 240, F(si + 2), M, 225, Kp);
      }
      for (int d = 0; d < 2; ++d) {
        int si = (pass ? QrB : PrB) + (L * 2 + d) * 4;
        k_gru_time<<<2, 512, 0, stream>>>(d ? GiB : GiA, rWhh[w0 + d], F(si + 3),
                                          U, 450, L * 150 + d * 75, X1, 160, d * 75, T, d);
      }
    }
  }

  // ---- QPP: precompute projections, then persistent gated recurrence ----
  k_pack_f16<<<(12800 * 480 + 255) / 256, 256, 0, stream>>>(uP, 450, 12800, 450, uPf16, 480, 12800);
  k_pack_f16<<<(1600 * 480 + 255) / 256, 256, 0, stream>>>(uQ, 450, 1600, 450, uQf16, 480, 1600);
  k_gemm_wmma<<<(800 * 5 + 7) / 8, 256, 0, stream>>>(uPf16, 480, pWPu, 480, uPWPu, 80, nullptr, 12800, 75, 480);
  k_gemm_wmma<<<(100 * 5 + 7) / 8, 256, 0, stream>>>(uQf16, 480, pWQu, 480, WQuQ, 80, nullptr, 1600, 75, 480);
  size_t smemQ = (size_t)(32 * 928 * 2) * 2 + 32 * 96 * 2 +
                 (size_t)(32 * 80 + 32 * 64 + 32 * 240 * 2) * 4;  // 204800 B
  k_qpp<<<1, 512, smemQ, stream>>>(uP, uPWPu, WQuQ, uQ, F(Iv), pWg1, pWPv,
                                   qWih[0], qWhh[0], F(QPPf + 2), F(QPPf + 3), vPbuf, 0, 0);
  k_qpp<<<1, 512, smemQ, stream>>>(uP, uPWPu, WQuQ, uQ, F(Iv), pWg1, pWPv,
                                   qWih[1], qWhh[1], F(QPPb + 2), F(QPPb + 3), vPbuf, 75, 1);

  // ---- self-matching ----
  k_pack_f16<<<(12800 * 160 + 255) / 256, 256, 0, stream>>>(vPbuf, 150, 12800, 150, vPf16, 160, 12800);
  k_gemm_wmma<<<(800 * 5 + 7) / 8, 256, 0, stream>>>(vPf16, 160, pSMWp, 160, WpvP, 80, nullptr, 12800, 75, 160);
  size_t smemS = (size_t)(32 * 320 * 2) * 2 + 32 * 96 * 2 +
                 (size_t)(32 * 80 + 32 * 240 * 2 + 32 * 400) * 4;  // 169984 B
  k_sm<<<1, 512, smemS, stream>>>(vPbuf, WpvP, F(Iv), pWg2, pSMWp2, smWih[0], smWhh[0],
                                  F(SMf + 2), F(SMf + 3), hPbuf, 0, 0);
  k_sm<<<1, 512, smemS, stream>>>(vPbuf, WpvP, F(Iv), pWg2, pSMWp2, smWih[1], smWhh[1],
                                  F(SMb + 2), F(SMb + 3), hPbuf, 75, 1);

  // ---- pointer network ----
  k_pack_f16<<<(12800 * 160 + 255) / 256, 256, 0, stream>>>(hPbuf, 150, 12800, 150, hPf16, 160, 12800);
  k_gemm_wmma<<<(800 * 5 + 7) / 8, 256, 0, stream>>>(hPf16, 160, pWPh, 160, WPhhP, 80, nullptr, 12800, 75, 160);
  size_t smemP = (size_t)(80 + 32 * 80 + 32 * 480 * 2 + 32 * 304 + 32 * 400) * 4;  // 223552 B
  k_pointer<<<1, 512, smemP, stream>>>(WQuQ, uQ, hPbuf, WPhhP, F(Iv), F(IWQv), F(IVQr),
                                       F(IWah), F(PNI + 0), F(PNI + 1), F(PNI + 2),
                                       F(PNI + 3), (float*)d_out);
}
